// TransformerDecoder_45037027065975
// MI455X (gfx1250) — matrix operand order repaired and checked by reading, not compiled, not run on hardware
//
#include <hip/hip_runtime.h>
#include <hip/hip_bf16.h>
#include <stdint.h>

// ---------------------------------------------------------------------------
// Problem constants (from the reference)
// ---------------------------------------------------------------------------
#define HIDN   512
#define SEQ    1024
#define NH     8
#define HHID   (NH * HIDN)     // 4096
#define NEXP   8
#define FFND   1024
#define NLAYER 2
#define LN_EPS 1e-5f

// ---------------------------------------------------------------------------
// CDNA5 WMMA types
// ---------------------------------------------------------------------------
typedef __bf16 bf16_t;
typedef bf16_t bf16x16 __attribute__((ext_vector_type(16)));
typedef float  f32x8   __attribute__((ext_vector_type(8)));
typedef float  f32x4   __attribute__((ext_vector_type(4)));
typedef unsigned int u32x2 __attribute__((ext_vector_type(2)));
typedef unsigned int u32x4 __attribute__((ext_vector_type(4)));

union FragU {
  unsigned short u[16];
  u32x4          q[2];
  bf16x16        v;
};

__device__ __forceinline__ unsigned short f32_to_bf16_rne(float f) {
  union { float f; unsigned int u; } cv;
  cv.f = f;
  const unsigned int u = cv.u;
  const unsigned int r = u + 0x7FFFu + ((u >> 16) & 1u);  // round-to-nearest-even
  return (unsigned short)(r >> 16);
}

__device__ __forceinline__ unsigned int pack_bf16x2(float lo, float hi) {
  return (unsigned int)f32_to_bf16_rne(lo) |
         ((unsigned int)f32_to_bf16_rne(hi) << 16);
}

// ---------------------------------------------------------------------------
// Generic batched GEMM:  C[z] = act( A[z] * B[z] + bias )
//   block tile 128(M) x 128(N), 8 waves (4x2), wave tile 32x64 = 2x4 WMMA
//   WMMA f32_16x16x32_bf16, f32 inputs rounded RNE to bf16 in LDS
//   transB: B_gemm[k][n] = B[n*ldb + k]  (used for Q*K^T)
//   C offset per batch z:  (z / cDiv)*cOuter + (z % cDiv)*cInner
// ---------------------------------------------------------------------------
#define BLK_M 128
#define BLK_N 128
#define BLK_K 32
#define LDS_STRIDE 40   // ushorts per row (80 B: 16B-aligned, bank-conflict-free)

__global__ __launch_bounds__(256)
void gemm_bf16_wmma(const float* __restrict__ A, const float* __restrict__ B,
                    const float* __restrict__ bias, float* __restrict__ C,
                    int M, int N, int K,
                    int lda, int ldb, int ldc,
                    long long aBatch, long long bBatch,
                    long long cOuter, long long cInner, int cDiv,
                    int transB, int relu) {
  // A tile: [128 rows][32 k]   bf16, row stride 40 ushorts
  // B tile: stored transposed  [128 n][32 k] bf16, row stride 40 ushorts
  __shared__ __align__(16) unsigned short As[BLK_M * LDS_STRIDE];
  __shared__ __align__(16) unsigned short BsT[BLK_N * LDS_STRIDE];

  const int tid    = threadIdx.x;
  const int z      = blockIdx.z;
  const int blockM = blockIdx.y * BLK_M;
  const int blockN = blockIdx.x * BLK_N;

  const float* Ab = A + (long long)z * aBatch;
  const float* Bb = B + (long long)z * bBatch;
  float*       Cb = C + (long long)(z / cDiv) * cOuter
                      + (long long)(z % cDiv) * cInner;

  const int wave  = tid >> 5;
  const int lane  = tid & 31;
  const int hi    = lane >> 4;   // lane half (0/1)
  const int l16   = lane & 15;
  const int waveM = wave >> 1;   // 0..3  -> M offset 32*waveM
  const int waveN = wave & 1;    // 0..1  -> N offset 64*waveN

  f32x8 acc[2][4];
#pragma unroll
  for (int i = 0; i < 2; ++i)
#pragma unroll
    for (int j = 0; j < 4; ++j)
#pragma unroll
      for (int r = 0; r < 8; ++r) acc[i][j][r] = 0.0f;

  for (int kk = 0; kk < K; kk += BLK_K) {
    __syncthreads();

    // ---- stage A tile (128 x 32): float4 global loads, b64 packed stores --
    {
      const int kq = (tid & 7) * 4;   // k quad 0..28
      const int r0 = tid >> 3;        // 0..31
#pragma unroll
      for (int i = 0; i < 4; ++i) {
        const int row = r0 + i * 32;
        const f32x4 v = *(const f32x4*)(Ab + (long long)(blockM + row) * lda + kk + kq);
        u32x2 p;
        p[0] = pack_bf16x2(v[0], v[1]);
        p[1] = pack_bf16x2(v[2], v[3]);
        *(u32x2*)((char*)As + row * (LDS_STRIDE * 2) + kq * 2) = p;
      }
      // prefetch next K tile of A (gfx1250 global_prefetch_b8)
      if (kk + BLK_K < K)
        __builtin_prefetch(Ab + (long long)(blockM + r0) * lda + kk + BLK_K + kq, 0, 1);
    }

    // ---- stage B tile as BsT[n][k] (128 n x 32 k) ----
    if (!transB) {
      // B row-major [K,N]: 4x4 register micro-transpose
      const int kb = (tid & 7) * 4;   // 0..28
      const int nb = (tid >> 3) * 4;  // 0..124
      f32x4 r0 = *(const f32x4*)(Bb + (long long)(kk + kb + 0) * ldb + blockN + nb);
      f32x4 r1 = *(const f32x4*)(Bb + (long long)(kk + kb + 1) * ldb + blockN + nb);
      f32x4 r2 = *(const f32x4*)(Bb + (long long)(kk + kb + 2) * ldb + blockN + nb);
      f32x4 r3 = *(const f32x4*)(Bb + (long long)(kk + kb + 3) * ldb + blockN + nb);
#pragma unroll
      for (int j = 0; j < 4; ++j) {
        u32x2 p;
        p[0] = pack_bf16x2(r0[j], r1[j]);
        p[1] = pack_bf16x2(r2[j], r3[j]);
        *(u32x2*)((char*)BsT + (nb + j) * (LDS_STRIDE * 2) + kb * 2) = p;
      }
      if (kk + BLK_K < K)
        __builtin_prefetch(Bb + (long long)(kk + BLK_K + kb) * ldb + blockN + nb, 0, 1);
    } else {
      // B row-major [N,K]: rows map directly to BsT rows
      const int kq = (tid & 7) * 4;   // 0..28
      const int n0 = tid >> 3;        // 0..31
#pragma unroll
      for (int i = 0; i < 4; ++i) {
        const int n = n0 + i * 32;
        const f32x4 v = *(const f32x4*)(Bb + (long long)(blockN + n) * ldb + kk + kq);
        u32x2 p;
        p[0] = pack_bf16x2(v[0], v[1]);
        p[1] = pack_bf16x2(v[2], v[3]);
        *(u32x2*)((char*)BsT + n * (LDS_STRIDE * 2) + kq * 2) = p;
      }
      if (kk + BLK_K < K)
        __builtin_prefetch(Bb + (long long)(blockN + n0) * ldb + kk + BLK_K + kq, 0, 1);
    }
    __syncthreads();

    // ---- fragments via ds_load_b128 (ISA 7.12.2 layouts) ----
    // A frag lane: M = lane%16; k = {8*hi+0..7, 16+8*hi+0..7} -> 2 contiguous runs
    FragU a[2];
#pragma unroll
    for (int ti = 0; ti < 2; ++ti) {
      const int row = waveM * 32 + ti * 16 + l16;
      const char* base = (const char*)As + row * (LDS_STRIDE * 2) + 16 * hi;
      a[ti].q[0] = *(const u32x4*)(base);
      a[ti].q[1] = *(const u32x4*)(base + 32);
    }
    // B frag lane: N = lane%16; k = {8*hi+0..7, 16+8*hi+0..7} -> 2 contiguous runs
    FragU b[4];
#pragma unroll
    for (int tj = 0; tj < 4; ++tj) {
      const int col = waveN * 64 + tj * 16 + l16;
      const char* base = (const char*)BsT + col * (LDS_STRIDE * 2) + 16 * hi;
      b[tj].q[0] = *(const u32x4*)(base);
      b[tj].q[1] = *(const u32x4*)(base + 32);
    }

#pragma unroll
    for (int ti = 0; ti < 2; ++ti)
#pragma unroll
      for (int tj = 0; tj < 4; ++tj)
        acc[ti][tj] = __builtin_amdgcn_wmma_f32_16x16x32_bf16(
            false, a[ti].v, false, b[tj].v, (short)0, acc[ti][tj],
            false, false);
  }

  // ---- epilogue: C 16x16 f32 layout: m = r + 8*half, n = lane%16 ----
#pragma unroll
  for (int ti = 0; ti < 2; ++ti) {
#pragma unroll
    for (int tj = 0; tj < 4; ++tj) {
      const int n = blockN + waveN * 64 + tj * 16 + l16;
      const float bvv = bias ? bias[n] : 0.0f;
#pragma unroll
      for (int r = 0; r < 8; ++r) {
        const int m = blockM + waveM * 32 + ti * 16 + hi * 8 + r;
        float v = acc[ti][tj][r] + bvv;
        if (relu) v = v > 0.0f ? v : 0.0f;
        Cb[(long long)m * ldc + n] = v;
      }
    }
  }
}

// ---------------------------------------------------------------------------
// Row softmax with optional causal mask (in-place).  One block per row.
// ---------------------------------------------------------------------------
__global__ __launch_bounds__(256)
void softmax_rows(float* __restrict__ sc, int S, int causal) {
  __shared__ float red[256];
  const long long row = blockIdx.x;
  const int q   = (int)(row % S);
  float* p      = sc + row * (long long)S;
  const int tid = threadIdx.x;

  float mx = -3.4e38f;
  for (int i = tid; i < S; i += 256) {
    float v = p[i];
    if (causal && i > q) v = -3.4e38f;
    mx = fmaxf(mx, v);
  }
  red[tid] = mx; __syncthreads();
  for (int s = 128; s > 0; s >>= 1) {
    if (tid < s) red[tid] = fmaxf(red[tid], red[tid + s]);
    __syncthreads();
  }
  mx = red[0]; __syncthreads();

  float sum = 0.0f;
  for (int i = tid; i < S; i += 256) {
    const float v = (causal && i > q) ? 0.0f : __expf(p[i] - mx);
    p[i] = v;
    sum += v;
  }
  red[tid] = sum; __syncthreads();
  for (int s = 128; s > 0; s >>= 1) {
    if (tid < s) red[tid] += red[tid + s];
    __syncthreads();
  }
  const float inv = 1.0f / red[0];
  for (int i = tid; i < S; i += 256) p[i] *= inv;
}

// ---------------------------------------------------------------------------
// dst = LayerNorm(xres + a * rowscale) * g + b      (one block per row)
// ---------------------------------------------------------------------------
__global__ __launch_bounds__(256)
void add_layernorm(const float* __restrict__ a, const float* __restrict__ xr,
                   const float* __restrict__ g, const float* __restrict__ b,
                   const float* __restrict__ rowscale,
                   float* __restrict__ dst) {
  __shared__ float sv[HIDN];
  __shared__ float red[256];
  const long long row = blockIdx.x;
  const int tid = threadIdx.x;
  const float scale = rowscale ? rowscale[row] : 1.0f;

  float local = 0.0f;
  for (int i = tid; i < HIDN; i += 256) {
    const float v = xr[row * HIDN + i] + a[row * HIDN + i] * scale;
    sv[i] = v;
    local += v;
  }
  red[tid] = local; __syncthreads();
  for (int s = 128; s > 0; s >>= 1) {
    if (tid < s) red[tid] += red[tid + s];
    __syncthreads();
  }
  const float mean = red[0] / (float)HIDN;
  __syncthreads();

  local = 0.0f;
  for (int i = tid; i < HIDN; i += 256) {
    const float d = sv[i] - mean;
    local += d * d;
  }
  red[tid] = local; __syncthreads();
  for (int s = 128; s > 0; s >>= 1) {
    if (tid < s) red[tid] += red[tid + s];
    __syncthreads();
  }
  const float rstd = rsqrtf(red[0] / (float)HIDN + LN_EPS);

  for (int i = tid; i < HIDN; i += 256)
    dst[row * HIDN + i] = g[i] * (sv[i] - mean) * rstd + b[i];
}

// ---------------------------------------------------------------------------
// MoE gating: logits = x @ gw + gb (E=8); softmax top-2; keep only k=TOPK-1
// (faithful "overwrite" bug): sel = 2nd-best index,
// wsel = p2/(p1+p2) = 1/(1+exp(l1-l2)).      One block (128 thr) per token.
// ---------------------------------------------------------------------------
__global__ __launch_bounds__(128)
void moe_gate(const float* __restrict__ x, const float* __restrict__ gw,
              const float* __restrict__ gb, int* __restrict__ sel,
              float* __restrict__ wsel) {
  __shared__ float red[128];
  __shared__ float logits[NEXP];
  const long long t = blockIdx.x;
  const int tid = threadIdx.x;

  float part[NEXP];
#pragma unroll
  for (int e = 0; e < NEXP; ++e) part[e] = 0.0f;
  for (int d = tid; d < HIDN; d += 128) {
    const float xv = x[t * HIDN + d];
#pragma unroll
    for (int e = 0; e < NEXP; ++e) part[e] += xv * gw[d * NEXP + e];
  }
  for (int e = 0; e < NEXP; ++e) {
    red[tid] = part[e]; __syncthreads();
    for (int s = 64; s > 0; s >>= 1) {
      if (tid < s) red[tid] += red[tid + s];
      __syncthreads();
    }
    if (tid == 0) logits[e] = red[0] + gb[e];
    __syncthreads();
  }
  if (tid == 0) {
    float l1 = -3.4e38f, l2 = -3.4e38f;
    int   i1 = 0,        i2 = 0;
    for (int e = 0; e < NEXP; ++e) {
      const float v = logits[e];
      if (v > l1) { l2 = l1; i2 = i1; l1 = v; i1 = e; }
      else if (v > l2) { l2 = v; i2 = e; }
    }
    (void)i1;
    sel[t]  = i2;
    wsel[t] = 1.0f / (1.0f + __expf(l1 - l2));
  }
}

// ---------------------------------------------------------------------------
// y = (init ? 0 : y) + (sel[token]==e ? f : 0)
// ---------------------------------------------------------------------------
__global__ __launch_bounds__(256)
void moe_accum(const float* __restrict__ f, const int* __restrict__ sel,
               float* __restrict__ y, int e, int init, long long total) {
  const long long idx = (long long)blockIdx.x * 256 + threadIdx.x;
  if (idx >= total) return;
  const long long t = idx / HIDN;
  const float v = (sel[t] == e) ? f[idx] : 0.0f;
  y[idx] = init ? v : (y[idx] + v);
}

// ---------------------------------------------------------------------------
// Host orchestration
// ---------------------------------------------------------------------------
extern "C" void kernel_launch(void* const* d_in, const int* in_sizes, int n_in,
                              void* d_out, int out_size, void* d_ws,
                              size_t ws_size, hipStream_t stream) {
  (void)n_in; (void)out_size; (void)ws_size;

  const float* in_x  = (const float*)d_in[0];
  const float* in_en = (const float*)d_in[1];
  const float* wq    = (const float*)d_in[2];
  const float* bq    = (const float*)d_in[3];
  const float* wk    = (const float*)d_in[4];
  const float* bk    = (const float*)d_in[5];
  const float* wv    = (const float*)d_in[6];
  const float* bv    = (const float*)d_in[7];
  const float* wo    = (const float*)d_in[8];
  const float* bo    = (const float*)d_in[9];
  const float* gw    = (const float*)d_in[10];
  const float* gb    = (const float*)d_in[11];
  const float* w1    = (const float*)d_in[12];
  const float* b1    = (const float*)d_in[13];
  const float* w2    = (const float*)d_in[14];
  const float* b2    = (const float*)d_in[15];
  const float* lng   = (const float*)d_in[16];
  const float* lnb   = (const float*)d_in[17];

  const int Bsz        = in_sizes[0] / (SEQ * HIDN);
  const long long nTok = (long long)Bsz * SEQ;            // 4096
  const long long szX  = nTok * HIDN;                     // 2M
  const long long szQ  = nTok * HHID;                     // 16.7M
  const long long szSC = (long long)Bsz * NH * SEQ * SEQ; // 33.5M

  float* W   = (float*)d_ws;
  float* X   = W; W += szX;
  float* Q   = W; W += szQ;
  float* Kb  = W; W += szQ;
  float* Vb  = W; W += szQ;
  float* SC  = W; W += szSC;
  float* HB  = W; W += nTok * FFND;
  float* FB  = W; W += szX;
  float* YB  = W; W += szX;
  float* WSL = W; W += nTok;
  int*   SEL = (int*)W;

  hipMemcpyAsync(X, in_x, (size_t)szX * sizeof(float),
                 hipMemcpyDeviceToDevice, stream);

  auto gemm = [&](const float* A, const float* Bm, const float* bias, float* C,
                  int M, int N, int K, int lda, int ldb, int ldc,
                  long long aB, long long bB,
                  long long cOuter, long long cInner, int cDiv,
                  int nb, int transB, int relu) {
    dim3 grid((unsigned)(N / BLK_N), (unsigned)(M / BLK_M), (unsigned)nb);
    gemm_bf16_wmma<<<grid, 256, 0, stream>>>(A, Bm, bias, C, M, N, K,
                                             lda, ldb, ldc, aB, bB,
                                             cOuter, cInner, cDiv,
                                             transB, relu);
  };

  for (int l = 0; l < NLAYER; ++l) {
    // -------- two attentions: 0 = masked self, 1 = cross --------
    for (int at = 0; at < 2; ++at) {
      const long long wi = (long long)(l * 2 + at);
      const float* qin = (at == 0) ? X : in_en;
      const float* kin = (at == 0) ? X : in_en;
      const float* vin = X;  // cross-attn values come from current x (faithful)

      // Q/K/V projections: [nTok,HID] @ [HID,HHID] + bias
      gemm(qin, wq + wi * HIDN * HHID, bq + wi * HHID, Q,
           (int)nTok, HHID, HIDN, HIDN, HHID, HHID, 0, 0, 0, 0, 1, 1, 0, 0);
      gemm(kin, wk + wi * HIDN * HHID, bk + wi * HHID, Kb,
           (int)nTok, HHID, HIDN, HIDN, HHID, HHID, 0, 0, 0, 0, 1, 1, 0, 0);
      gemm(vin, wv + wi * HIDN * HHID, bv + wi * HHID, Vb,
           (int)nTok, HHID, HIDN, HIDN, HHID, HHID, 0, 0, 0, 0, 1, 1, 0, 0);

      // scores[z] = Q[z] @ K[z]^T, z over B*H (scale = 1.0 per reference)
      gemm(Q, Kb, nullptr, SC, SEQ, SEQ, HIDN, HIDN, HIDN, SEQ,
           (long long)SEQ * HIDN, (long long)SEQ * HIDN,
           (long long)SEQ * SEQ, 0, 1, Bsz * NH, 1, 0);

      softmax_rows<<<dim3((unsigned)(Bsz * NH * SEQ)), 256, 0, stream>>>(
          SC, SEQ, at == 0 ? 1 : 0);

      // attn @ V, written head-transposed into Kb as [B,S,H*HID]
      gemm(SC, Vb, nullptr, Kb, SEQ, HIDN, SEQ, SEQ, HIDN, HHID,
           (long long)SEQ * SEQ, (long long)SEQ * HIDN,
           (long long)SEQ * HHID, (long long)HIDN, NH, Bsz * NH, 0, 0);

      // output projection: [nTok,HHID] @ [HHID,HID] + bo  -> FB
      gemm(Kb, wo + wi * HHID * HIDN, bo + wi * HIDN, FB,
           (int)nTok, HIDN, HHID, HHID, HIDN, HIDN, 0, 0, 0, 0, 1, 1, 0, 0);

      // x = LN(x + a)
      add_layernorm<<<dim3((unsigned)nTok), 256, 0, stream>>>(
          FB, X, lng + (long long)(l * 3 + at) * HIDN,
          lnb + (long long)(l * 3 + at) * HIDN, nullptr, X);
    }

    // -------- MoE --------
    moe_gate<<<dim3((unsigned)nTok), 128, 0, stream>>>(
        X, gw + (long long)l * HIDN * NEXP, gb + (long long)l * NEXP, SEL, WSL);

    for (int e = 0; e < NEXP; ++e) {
      const long long we = (long long)(l * NEXP + e);
      gemm(X, w1 + we * HIDN * FFND, b1 + we * FFND, HB,
           (int)nTok, FFND, HIDN, HIDN, FFND, FFND, 0, 0, 0, 0, 1, 1, 0, 1);
      gemm(HB, w2 + we * FFND * HIDN, b2 + we * HIDN, FB,
           (int)nTok, HIDN, FFND, FFND, HIDN, HIDN, 0, 0, 0, 0, 1, 1, 0, 0);
      moe_accum<<<dim3((unsigned)((szX + 255) / 256)), 256, 0, stream>>>(
          FB, SEL, YB, e, e == 0 ? 1 : 0, szX);
    }

    // x = LN(x + y * wsel) ; last layer writes straight to d_out
    float* dst = (l == NLAYER - 1) ? (float*)d_out : X;
    add_layernorm<<<dim3((unsigned)nTok), 256, 0, stream>>>(
        YB, X, lng + (long long)(l * 3 + 2) * HIDN,
        lnb + (long long)(l * 3 + 2) * HIDN, WSL, dst);
  }
}
